// EdgeLevelGNN_2147483648415
// MI455X (gfx1250) — matrix operand order repaired and checked by reading, not compiled, not run on hardware
//
#include <hip/hip_runtime.h>
#include <stdint.h>

// ---------------- CDNA5 (gfx1250) types ----------------
typedef __attribute__((ext_vector_type(16))) __bf16 v16bf;
typedef __attribute__((ext_vector_type(8)))  float  v8f;
typedef __attribute__((ext_vector_type(4)))  unsigned int v4u;
typedef __attribute__((ext_vector_type(8)))  int v8i;
typedef __attribute__((ext_vector_type(4)))  int v4i;

#define NN 50000
#define NE 800000
#define D  128
#define BN_EPS 1e-5f

// float -> bf16 round-to-nearest-even (bit twiddle; NaN -> qNaN)
__device__ __forceinline__ unsigned short f2bf(float f) {
  union { float f; unsigned u; } v; v.f = f;
  unsigned u = v.u;
  if ((u & 0x7fffffffu) > 0x7f800000u) return (unsigned short)0x7fc0;
  unsigned r = u + 0x7fffu + ((u >> 16) & 1u);
  return (unsigned short)(r >> 16);
}
__device__ __forceinline__ float bf2f(unsigned short b) {
  union { unsigned u; float f; } v; v.u = ((unsigned)b) << 16;
  return v.f;
}

// WMMA operand fragment: 16 bf16 across 2 dwordx4 loads
union Frag {
  uint4 q[2];
  unsigned short u[16];
  v16bf v;
};

// ---------------- TDM: 2D tile (whole matrix) global -> LDS ----------------
// D# per CDNA5 ISA: group0 = {flags, lds_addr, global_addr, type=2},
// group1 = {data_size=2B, tensor dims, tile dims, dim0 stride}. Groups 2/3 zero (2D).
// This toolchain exposes the 6-arg builtin: (g0, g1, g2, g3, g4, cpol).
__device__ __forceinline__ void tdm_load_2d(const void* gptr, unsigned lds_off,
                                            unsigned rowlen_elems, unsigned nrows) {
  unsigned long long ga = (unsigned long long)gptr;
  v4u g0;
  g0[0] = 1u;                                            // count=1, user descriptor
  g0[1] = lds_off;                                       // LDS byte address
  g0[2] = (unsigned)(ga & 0xffffffffu);                  // global_addr[31:0]
  g0[3] = (unsigned)((ga >> 32) & 0x01ffffffu) | 0x80000000u; // addr[56:32] | type=2
  v8i g1;
  g1[0] = (int)(1u << 16);                               // data_size=1 (2 bytes)
  g1[1] = (int)((rowlen_elems & 0xffffu) << 16);         // tensor_dim0[15:0] @ bit48
  g1[2] = (int)((rowlen_elems >> 16) | ((nrows & 0xffffu) << 16)); // dim0 hi | dim1 lo
  g1[3] = (int)((nrows >> 16) | ((rowlen_elems & 0xffffu) << 16)); // dim1 hi | tile_dim0
  g1[4] = (int)(nrows & 0xffffu);                        // tile_dim1 | tile_dim2=0
  g1[5] = (int)rowlen_elems;                             // tensor_dim0_stride lo32
  g1[6] = 0;                                             // stride hi | dim1_stride lo
  g1[7] = 0;
  v4i z4 = {0, 0, 0, 0};
  v8i z8 = {0, 0, 0, 0, 0, 0, 0, 0};
  __builtin_amdgcn_tensor_load_to_lds(g0, g1, z4, z4, z8, 0);
}

// ---------------- small prep kernels ----------------
__global__ void x_prep(const float* __restrict__ x, unsigned short* __restrict__ xbf, int total) {
  int i = blockIdx.x * blockDim.x + threadIdx.x;
  if (i >= total) return;
  union { float f; unsigned u; } v; v.f = x[i];
  unsigned a = v.u & 0x7fffffffu;
  float val = v.f;
  if (a >= 0x7f800000u) { // nan_to_num semantics
    if (a > 0x7f800000u) val = 0.0f;
    else val = (v.u >> 31) ? -3.4028235e38f : 3.4028235e38f;
  }
  xbf[i] = f2bf(val);
}

// Transpose-convert weight [K,N] fp32 -> [N,K] bf16 so B-fragments are contiguous per lane
__global__ void conv_wT(const float* __restrict__ W, unsigned short* __restrict__ WT, int K, int Ncols) {
  int i = blockIdx.x * blockDim.x + threadIdx.x;
  if (i >= K * Ncols) return;
  int k = i / Ncols, n = i - k * Ncols;
  WT[n * K + k] = f2bf(W[i]);
}

__global__ void zero_f32(float* __restrict__ p, int n) {
  int i = blockIdx.x * blockDim.x + threadIdx.x;
  if (i < n) p[i] = 0.0f;
}

// ---------------- degree / norm ----------------
__global__ void deg_init(float* __restrict__ deg, int n) {
  int i = blockIdx.x * blockDim.x + threadIdx.x;
  if (i < n) deg[i] = 1.0f; // self loop
}
__global__ void deg_scatter(const int* __restrict__ dst, float* __restrict__ deg, int ne) {
  int e = blockIdx.x * blockDim.x + threadIdx.x;
  if (e < ne) atomicAdd(&deg[dst[e]], 1.0f);
}
__global__ void deg_finish(float* __restrict__ deg, int n) {
  int i = blockIdx.x * blockDim.x + threadIdx.x;
  if (i < n) deg[i] = rsqrtf(deg[i]); // deg >= 1 always
}

// ---------------- node GEMM: tmp[N,128] = Abf[N,128] @ W[128,128]  (bf16 WMMA, f32 acc) ----------------
__global__ __launch_bounds__(256)
void gemm_node(const unsigned short* __restrict__ A,   // [N,128] bf16 row-major
               const unsigned short* __restrict__ WT,  // [128,128] bf16, [n][k]
               float* __restrict__ Cout, int nrows) {
  const int lane = threadIdx.x & 31;
  const int wave = threadIdx.x >> 5;
  const int r0 = blockIdx.x * 128 + wave * 16;
  int mrow = r0 + (lane & 15);
  if (mrow >= nrows) mrow = nrows - 1;         // clamp loads, stores guarded below
  const int koff = (lane >> 4) * 8;            // A: lanes 0-15 K{0..7,16..23}, 16-31 K{8..15,24..31}
  const int nb   = lane & 15;
  const int kb16 = (lane >> 4) * 8;            // B: lanes 0-15 K{0..7,16..23}, 16-31 K{8..15,24..31}

  v8f acc[8] = {};
  #pragma unroll
  for (int c = 0; c < 4; ++c) {
    Frag a;
    const unsigned short* ap = A + (size_t)mrow * D + c * 32 + koff;
    a.q[0] = *reinterpret_cast<const uint4*>(ap);
    a.q[1] = *reinterpret_cast<const uint4*>(ap + 16);
    #pragma unroll
    for (int n = 0; n < 8; ++n) {
      Frag b;
      const unsigned short* bp = WT + (size_t)(n * 16 + nb) * D + c * 32 + kb16;
      b.q[0] = *reinterpret_cast<const uint4*>(bp);
      b.q[1] = *reinterpret_cast<const uint4*>(bp + 16);
      acc[n] = __builtin_amdgcn_wmma_f32_16x16x32_bf16(false, a.v, false, b.v,
                                                       (short)0, acc[n], false, false);
    }
  }
  const int rowadd = (lane >> 4) * 8; // C layout: VGPR v -> row v (lanes<16) / 8+v (lanes>=16)
  #pragma unroll
  for (int n = 0; n < 8; ++n) {
    #pragma unroll
    for (int vI = 0; vI < 8; ++vI) {
      int row = r0 + rowadd + vI;
      if (row < nrows) Cout[(size_t)row * D + n * 16 + (lane & 15)] = acc[n][vI];
    }
  }
}

// ---------------- scatter-aggregate ----------------
// agg[i] = tmp[i] * dinv[i]^2   (self-loop contribution, also initializes agg)
__global__ void selfloop_init(const float* __restrict__ tmp, const float* __restrict__ dinv,
                              float* __restrict__ agg, int total) {
  int i = blockIdx.x * blockDim.x + threadIdx.x;
  if (i >= total) return;
  float w = dinv[i >> 7];
  agg[i] = tmp[i] * (w * w);
}

// agg[dst] += tmp[src] * dinv[src]*dinv[dst]; 32 threads/edge, 4 feats/thread
__global__ __launch_bounds__(256)
void scatter_edges(const float* __restrict__ tmp, const float* __restrict__ dinv,
                   const int* __restrict__ src, const int* __restrict__ dst,
                   float* __restrict__ agg) {
  long long t = (long long)blockIdx.x * blockDim.x + threadIdx.x;
  int e = (int)(t >> 5);
  if (e >= NE) return;
  int f0 = ((int)t & 31) * 4;
  int s = src[e], d2 = dst[e];
  float w = dinv[s] * dinv[d2];
  float4 m = *reinterpret_cast<const float4*>(tmp + (size_t)s * D + f0);
  float* base = agg + (size_t)d2 * D + f0;
  atomicAdd(base + 0, m.x * w);
  atomicAdd(base + 1, m.y * w);
  atomicAdd(base + 2, m.z * w);
  atomicAdd(base + 3, m.w * w);
}

// ---------------- batchnorm ----------------
// stats[0..127]=sum, stats[128..255]=sumsq; each thread owns one feature column
__global__ __launch_bounds__(256)
void bn_stats(const float* __restrict__ agg, float* __restrict__ stats, int nrows) {
  int f = threadIdx.x & 127;
  int row = blockIdx.x * 2 + (threadIdx.x >> 7);
  int stride = gridDim.x * 2;
  float s = 0.f, s2 = 0.f;
  for (; row < nrows; row += stride) {
    float v = agg[(size_t)row * D + f];
    s += v; s2 += v * v;
  }
  atomicAdd(&stats[f], s);
  atomicAdd(&stats[D + f], s2);
}

// h = relu(gamma*(x-mu)*rsqrt(var+eps)+beta) -> bf16  (conv bias cancels in BN mean)
__global__ void bn_apply(const float* __restrict__ agg, const float* __restrict__ stats,
                         const float* __restrict__ gamma, const float* __restrict__ beta,
                         unsigned short* __restrict__ outbf, int total) {
  int i = blockIdx.x * blockDim.x + threadIdx.x;
  if (i >= total) return;
  int f = i & 127;
  const float inv_n = 1.0f / (float)NN;
  float mu  = stats[f] * inv_n;
  float var = stats[D + f] * inv_n - mu * mu;
  float sc  = gamma[f] * rsqrtf(var + BN_EPS);
  float val = (agg[i] - mu) * sc + beta[f];
  outbf[i] = f2bf(fmaxf(val, 0.0f));
}

// ---------------- edge-level MLP head ----------------
// per wave: 16 edges -> gather [h[src]|h[dst]] (16x256 bf16) -> 128 -> 64 -> 2
// Weights staged once per block into LDS by the Tensor Data Mover.
__global__ __launch_bounds__(256)
void edge_head(const unsigned short* __restrict__ h,    // [N,128] bf16
               const int* __restrict__ src, const int* __restrict__ dst,
               const unsigned short* __restrict__ Wc1T, // [128][256] bf16
               const float* __restrict__ bc1,
               const unsigned short* __restrict__ Wc2T, // [64][128] bf16
               const float* __restrict__ bc2,
               const float* __restrict__ Wc3,           // [64,2] fp32
               const float* __restrict__ bc3,
               float* __restrict__ out) {
  __shared__ __align__(16) unsigned short w1s[128 * 256]; // 64KB, shared by all waves
  __shared__ __align__(16) unsigned short w2s[64 * 128];  // 16KB
  __shared__ __align__(16) unsigned short z1s[8][16 * 128]; // per-wave private (32KB)
  __shared__ __align__(16) unsigned short z2s[8][16 * 64];  // per-wave private (16KB)

  const int lane = threadIdx.x & 31;
  const int wave = threadIdx.x >> 5;

  // TDM: DMA both weight matrices into LDS (wave 0 issues; TDM ignores EXEC)
  if (wave == 0) {
    tdm_load_2d(Wc1T, (unsigned)(unsigned long long)(uintptr_t)&w1s[0], 256u, 128u);
    tdm_load_2d(Wc2T, (unsigned)(unsigned long long)(uintptr_t)&w2s[0], 128u, 64u);
    __builtin_amdgcn_s_wait_tensorcnt(0);
  }
  __syncthreads();

  const int e0 = (blockIdx.x * 8 + wave) * 16;
  const int er = lane & 15;
  const int koff = (lane >> 4) * 8;
  const int kb16base = (lane >> 4) * 8;
  const int myedge = e0 + er;
  const int nsrc = src[myedge], ndst = dst[myedge];

  // ---- layer 1: (16x256) @ (256x128), A gathered straight into fragment layout ----
  v8f acc1[8] = {};
  #pragma unroll
  for (int c = 0; c < 8; ++c) {
    int kk = c * 32 + koff; // chunk never straddles src/dst boundary (128 = chunk 4 start)
    const unsigned short* base = (kk < D) ? (h + (size_t)nsrc * D + kk)
                                          : (h + (size_t)ndst * D + (kk - D));
    Frag a;
    a.q[0] = *reinterpret_cast<const uint4*>(base);
    a.q[1] = *reinterpret_cast<const uint4*>(base + 16);
    int kb16 = c * 32 + kb16base;
    #pragma unroll
    for (int n = 0; n < 8; ++n) {
      Frag b;
      const unsigned short* bp = &w1s[(n * 16 + er) * 256 + kb16];
      b.q[0] = *reinterpret_cast<const uint4*>(bp);
      b.q[1] = *reinterpret_cast<const uint4*>(bp + 16);
      acc1[n] = __builtin_amdgcn_wmma_f32_16x16x32_bf16(false, a.v, false, b.v,
                                                        (short)0, acc1[n], false, false);
    }
  }
  // bias + relu, C-layout -> row-major bf16 in LDS (cross-lane re-layout for next A)
  const int rowadd = (lane >> 4) * 8;
  #pragma unroll
  for (int n = 0; n < 8; ++n) {
    int col = n * 16 + er;
    float bb = bc1[col];
    #pragma unroll
    for (int vI = 0; vI < 8; ++vI)
      z1s[wave][(rowadd + vI) * 128 + col] = f2bf(fmaxf(acc1[n][vI] + bb, 0.0f));
  }
  __builtin_amdgcn_wave_barrier(); // DS ops are in-order per wave; fence the compiler

  // ---- layer 2: (16x128) @ (128x64) ----
  v8f acc2[4] = {};
  #pragma unroll
  for (int c = 0; c < 4; ++c) {
    Frag a;
    const unsigned short* ap = &z1s[wave][er * 128 + c * 32 + koff];
    a.q[0] = *reinterpret_cast<const uint4*>(ap);
    a.q[1] = *reinterpret_cast<const uint4*>(ap + 16);
    int kb16 = c * 32 + kb16base;
    #pragma unroll
    for (int n = 0; n < 4; ++n) {
      Frag b;
      const unsigned short* bp = &w2s[(n * 16 + er) * 128 + kb16];
      b.q[0] = *reinterpret_cast<const uint4*>(bp);
      b.q[1] = *reinterpret_cast<const uint4*>(bp + 16);
      acc2[n] = __builtin_amdgcn_wmma_f32_16x16x32_bf16(false, a.v, false, b.v,
                                                        (short)0, acc2[n], false, false);
    }
  }
  #pragma unroll
  for (int n = 0; n < 4; ++n) {
    int col = n * 16 + er;
    float bb = bc2[col];
    #pragma unroll
    for (int vI = 0; vI < 8; ++vI)
      z2s[wave][(rowadd + vI) * 64 + col] = f2bf(fmaxf(acc2[n][vI] + bb, 0.0f));
  }
  __builtin_amdgcn_wave_barrier();

  // ---- layer 3: 64 -> 2; one lane per (edge, output) ----
  int e3 = lane >> 1, o = lane & 1;
  float s3 = bc3[o];
  #pragma unroll 8
  for (int k = 0; k < 64; ++k)
    s3 += bf2f(z2s[wave][e3 * 64 + k]) * Wc3[k * 2 + o];
  out[(size_t)(e0 + e3) * 2 + o] = s3;
}

// ---------------- host launch ----------------
extern "C" void kernel_launch(void* const* d_in, const int* in_sizes, int n_in,
                              void* d_out, int out_size, void* d_ws, size_t ws_size,
                              hipStream_t stream) {
  (void)in_sizes; (void)n_in; (void)out_size; (void)ws_size;
  const float* x   = (const float*)d_in[0];
  const int*   ei  = (const int*)d_in[1];
  const float* W1  = (const float*)d_in[2];
  const float* W2  = (const float*)d_in[4];
  const float* W3  = (const float*)d_in[6];
  const float* g1  = (const float*)d_in[8];  const float* be1 = (const float*)d_in[9];
  const float* g2  = (const float*)d_in[10]; const float* be2 = (const float*)d_in[11];
  const float* g3  = (const float*)d_in[12]; const float* be3 = (const float*)d_in[13];
  const float* Wc1 = (const float*)d_in[14]; const float* bc1 = (const float*)d_in[15];
  const float* Wc2 = (const float*)d_in[16]; const float* bc2 = (const float*)d_in[17];
  const float* Wc3 = (const float*)d_in[18]; const float* bc3 = (const float*)d_in[19];
  const int* src = ei;
  const int* dst = ei + NE;
  float* out = (float*)d_out;

  // workspace carve (bump allocator, 256B aligned)
  char* p = (char*)d_ws;
  auto alloc = [&](size_t bytes) -> char* {
    char* r = p; p += (bytes + 255) & ~(size_t)255; return r;
  };
  unsigned short* actbf = (unsigned short*)alloc(2ull * NN * D);   // bf16 activations (in/out)
  float* tmp   = (float*)alloc(4ull * NN * D);                     // GEMM output
  float* agg   = (float*)alloc(4ull * NN * D);                     // scatter accumulator
  float* dinv  = (float*)alloc(4ull * NN);
  float* stats = (float*)alloc(4ull * 256);
  unsigned short* W1T  = (unsigned short*)alloc(2ull * 128 * 128);
  unsigned short* W2T  = (unsigned short*)alloc(2ull * 128 * 128);
  unsigned short* W3T  = (unsigned short*)alloc(2ull * 128 * 128);
  unsigned short* Wc1T = (unsigned short*)alloc(2ull * 256 * 128);
  unsigned short* Wc2T = (unsigned short*)alloc(2ull * 128 * 64);

  const int T = 256;
  const int totND = NN * D;

  // prep: bf16 activations + transposed bf16 weights
  x_prep<<<(totND + T - 1) / T, T, 0, stream>>>(x, actbf, totND);
  conv_wT<<<(128 * 128 + T - 1) / T, T, 0, stream>>>(W1, W1T, 128, 128);
  conv_wT<<<(128 * 128 + T - 1) / T, T, 0, stream>>>(W2, W2T, 128, 128);
  conv_wT<<<(128 * 128 + T - 1) / T, T, 0, stream>>>(W3, W3T, 128, 128);
  conv_wT<<<(256 * 128 + T - 1) / T, T, 0, stream>>>(Wc1, Wc1T, 256, 128);
  conv_wT<<<(128 * 64  + T - 1) / T, T, 0, stream>>>(Wc2, Wc2T, 128, 64);

  // degree -> dinv = deg^-1/2 (with self loops)
  deg_init<<<(NN + T - 1) / T, T, 0, stream>>>(dinv, NN);
  deg_scatter<<<(NE + T - 1) / T, T, 0, stream>>>(dst, dinv, NE);
  deg_finish<<<(NN + T - 1) / T, T, 0, stream>>>(dinv, NN);

  const int gemmBlocks = (NN + 127) / 128;
  const int scatBlocks = (int)(((long long)NE * 32 + T - 1) / T);
  const unsigned short* WTs[3] = {W1T, W2T, W3T};
  const float* gs[3]  = {g1, g2, g3};
  const float* bes[3] = {be1, be2, be3};

  for (int L = 0; L < 3; ++L) {
    gemm_node<<<gemmBlocks, T, 0, stream>>>(actbf, WTs[L], tmp, NN);
    selfloop_init<<<(totND + T - 1) / T, T, 0, stream>>>(tmp, dinv, agg, totND);
    scatter_edges<<<scatBlocks, T, 0, stream>>>(tmp, dinv, src, dst, agg);
    zero_f32<<<1, 256, 0, stream>>>(stats, 256);
    bn_stats<<<400, T, 0, stream>>>(agg, stats, NN);
    bn_apply<<<(totND + T - 1) / T, T, 0, stream>>>(agg, stats, gs[L], bes[L], actbf, totND);
  }

  // edge-level MLP head: 8 waves/block, 16 edges/wave -> 128 edges/block; 800000/128 = 6250 exact
  edge_head<<<NE / 128, T, 0, stream>>>(actbf, src, dst, Wc1T, bc1, Wc2T, bc2, Wc3, bc3, out);
}
